// MultiScaleRetention_58265526338092
// MI455X (gfx1250) — compile-verified, matrix operand order checked by reading, not run on hardware
//
#include <hip/hip_runtime.h>

// ---------------------------------------------------------------------------
// MultiScaleRetention for MI455X (gfx1250, wave32, WMMA bf16 16x16x32 + TDM).
//
// Pipeline (all on `stream`):
//   1. cast_f32_bf16_v4: x, wq, wk, wv, wo -> bf16 copies in d_ws
//   2. gemm_bf16 (x3): q = (x@Wq^T)*0.125 -> [B,H,L,64] bf16
//                      k =  x@Wk^T        -> [B,H,L,64] bf16
//                      v =  x@Wv^T        -> [B,H,64,L] bf16 (transposed)
//      B tiles staged in LDS via TENSOR_LOAD_TO_LDS (double buffered),
//      shared by all 8 waves of the block.
//   3. retention: fused causal-decay attention, 16x64 fp32 accum per wave.
//   4. gemm_bf16: out = retained @ Wo^T -> fp32 to d_out
// ---------------------------------------------------------------------------

typedef unsigned short u16;
typedef __attribute__((ext_vector_type(16))) __bf16 v16bf;
typedef __attribute__((ext_vector_type(8)))  float  v8f;
typedef __attribute__((ext_vector_type(4)))  unsigned int u32x4;
typedef __attribute__((ext_vector_type(8)))  int i32x8;
typedef __attribute__((ext_vector_type(4)))  int i32x4;

#define B_SZ  2
#define L_SZ  2048
#define D_SZ  1024
#define H_SZ  16
#define HD_SZ 64
#define M_SZ  (B_SZ * L_SZ)   // 4096 rows

#if defined(__has_builtin)
#if __has_builtin(__builtin_amdgcn_tensor_load_to_lds)
#define HAVE_TDM 1
#endif
#endif
#ifndef HAVE_TDM
#define HAVE_TDM 0
#endif

union FragU { v16bf v; uint4 u[2]; };

__device__ __forceinline__ u16 f2bf(float x) {
  union { float f; unsigned u; } c; c.f = x;
  unsigned r = (c.u + 0x7FFFu + ((c.u >> 16) & 1u)) >> 16;  // RNE
  return (u16)r;
}

// 16x32 bf16 fragment from row-major memory, leading dim `ld` (elements).
// 16-bit A-layout: lanes 0-15 (row m) hold K = {0..7,16..23},
// lanes 16-31 hold K = {8..15,24..31} -> two b128 loads per lane.
__device__ __forceinline__ v16bf load_frag(const u16* __restrict__ base,
                                           unsigned ld, unsigned row0, unsigned k0) {
  unsigned lane = threadIdx.x & 31u;
  unsigned off = (row0 + (lane & 15u)) * ld + k0 + ((lane >> 4) << 3);
  FragU f;
  f.u[0] = *(const uint4*)(base + off);
  f.u[1] = *(const uint4*)(base + off + 16);
  return f.v;
}

// Same fragment pattern read from an LDS tile with row stride 32 elements.
__device__ __forceinline__ v16bf lds_frag(const u16* buf, unsigned row0) {
  unsigned lane = threadIdx.x & 31u;
  const u16* p = buf + (row0 + (lane & 15u)) * 32u + ((lane >> 4) << 3);
  FragU f;
  f.u[0] = *(const uint4*)(p);
  f.u[1] = *(const uint4*)(p + 16);
  return f.v;
}

__device__ __forceinline__ v8f wmma_bf16(v16bf a, v16bf b, v8f c) {
  return __builtin_amdgcn_wmma_f32_16x16x32_bf16(false, a, false, b,
                                                 (short)0, c, false, false);
}

__device__ __forceinline__ void wait_tensorcnt0() {
#if defined(__has_builtin) && __has_builtin(__builtin_amdgcn_s_wait_tensorcnt)
  __builtin_amdgcn_s_wait_tensorcnt(0);
#else
  asm volatile("s_wait_tensorcnt 0" ::: "memory");
#endif
}

// Issue a TDM 2-D tile load: tile_d1 rows x tile_d0 elements of 2-byte data,
// row stride `stride_d0` elements, packed contiguously at LDS byte `lds_off`.
__device__ __forceinline__ void tdm_load_tile_2d(unsigned lds_off, const void* gptr,
                                                 unsigned tile_d0, unsigned tile_d1,
                                                 unsigned stride_d0) {
#if HAVE_TDM
  unsigned long long ga = (unsigned long long)(size_t)gptr;
  u32x4 g0 = {0u, 0u, 0u, 0u};
  g0.x = 1u;                                    // count=1 (valid descriptor)
  g0.y = lds_off;                               // lds_addr (bytes)
  g0.z = (unsigned)ga;                          // global_addr[31:0]
  g0.w = ((unsigned)(ga >> 32) & 0x01FFFFFFu)   // global_addr[56:32]
         | (2u << 30);                          // type = 2 ("image")
  i32x8 g1;
  g1[0] = (int)(1u << 16);                      // data_size = 1 -> 2 bytes
  g1[1] = (int)((tile_d0 & 0xFFFFu) << 16);     // tensor_dim0[15:0] = tile_d0
  g1[2] = (int)(((tile_d0 >> 16) & 0xFFFFu) |
                ((tile_d1 & 0xFFFFu) << 16));   // tensor_dim0 hi | tensor_dim1 lo
  g1[3] = (int)(((tile_d1 >> 16) & 0xFFFFu) |
                (tile_d0 << 16));               // tensor_dim1 hi | tile_dim0
  g1[4] = (int)(tile_d1 & 0xFFFFu);             // tile_dim1 (tile_dim2 = 0)
  g1[5] = (int)stride_d0;                       // tensor_dim0_stride[31:0]
  g1[6] = 0;                                    // stride hi / dim1_stride lo
  g1[7] = 0;
  i32x4 z4 = {0, 0, 0, 0};
#if __clang_major__ >= 23
  i32x8 z8 = {0, 0, 0, 0, 0, 0, 0, 0};
  __builtin_amdgcn_tensor_load_to_lds(g0, g1, z4, z4, z8, 0);
#else
  __builtin_amdgcn_tensor_load_to_lds(g0, g1, z4, z4, 0);
#endif
#else
  (void)lds_off; (void)gptr; (void)tile_d0; (void)tile_d1; (void)stride_d0;
#endif
}

// ---------------------------------------------------------------------------
__global__ void cast_f32_bf16_v4(const float* __restrict__ in,
                                 u16* __restrict__ out, int n4) {
  int i = blockIdx.x * blockDim.x + threadIdx.x;
  if (i < n4) {
    float4 f = ((const float4*)in)[i];
    ushort4 o;
    o.x = f2bf(f.x); o.y = f2bf(f.y); o.z = f2bf(f.z); o.w = f2bf(f.w);
    ((ushort4*)out)[i] = o;
  }
}

// ---------------------------------------------------------------------------
// C = scale * (A[M,K] @ W[N,K]^T)
// mode 0: fp32 row-major [M,N] to outF
// mode 2: bf16 to outB laid out [B, H, L, HD]   (for q, k)
// mode 3: bf16 to outB laid out [B, H, HD, L]   (transposed, for v)
// Block = 256 threads (8 waves); block tile 128x64; wave tile 16x64.
// The 64x32 B tile per k-step is staged in LDS (TDM, double buffered) and
// shared by all 8 waves.
__global__ void gemm_bf16(const u16* __restrict__ A, const u16* __restrict__ W,
                          float* __restrict__ outF, u16* __restrict__ outB,
                          int M, int N, int K, float scale, int mode) {
  __shared__ __align__(16) u16 bsm[2][64 * 32];   // 2 x 4KB B tiles

  int tid  = threadIdx.x;
  int w    = tid >> 5;
  int lane = tid & 31;
  unsigned m0 = blockIdx.x * 128 + w * 16;
  unsigned n0 = blockIdx.y * 64;
  int nsteps = K / 32;

#if HAVE_TDM
  if (tid < 32) {
    tdm_load_tile_2d((unsigned)(size_t)&bsm[0][0], W + (size_t)n0 * K,
                     32u, 64u, (unsigned)K);
  }
#else
  {  // cooperative copy of tile 0: 64 rows x 64B, one uint4 per thread
    int row = tid >> 2, c4 = tid & 3;
    ((uint4*)&bsm[0][row * 32])[c4] =
        *(const uint4*)(W + (size_t)(n0 + row) * K + c4 * 8);
  }
#endif

  v8f acc0 = {}, acc1 = {}, acc2 = {}, acc3 = {};
  for (int i = 0; i < nsteps; ++i) {
    unsigned k = (unsigned)i * 32u;
#if HAVE_TDM
    if (tid < 32) wait_tensorcnt0();
    __syncthreads();
    if (i + 1 < nsteps && tid < 32) {
      tdm_load_tile_2d((unsigned)(size_t)&bsm[(i + 1) & 1][0],
                       W + (size_t)n0 * K + k + 32u, 32u, 64u, (unsigned)K);
    }
#else
    __syncthreads();
    if (i + 1 < nsteps) {
      int row = tid >> 2, c4 = tid & 3;
      ((uint4*)&bsm[(i + 1) & 1][row * 32])[c4] =
          *(const uint4*)(W + (size_t)(n0 + row) * K + k + 32u + c4 * 8);
    }
#endif
    const u16* bt = bsm[i & 1];
    v16bf a = load_frag(A, (unsigned)K, m0, k);
    acc0 = wmma_bf16(a, lds_frag(bt,  0), acc0);
    acc1 = wmma_bf16(a, lds_frag(bt, 16), acc1);
    acc2 = wmma_bf16(a, lds_frag(bt, 32), acc2);
    acc3 = wmma_bf16(a, lds_frag(bt, 48), acc3);
  }

  // C/D layout: lane l, VGPR j -> (row = j + 8*(l>>4), col = l&15)
  int hi = lane >> 4, col = lane & 15;
  v8f accs[4] = {acc0, acc1, acc2, acc3};
#pragma unroll
  for (int nt = 0; nt < 4; ++nt) {
#pragma unroll
    for (int j = 0; j < 8; ++j) {
      float val = accs[nt][j] * scale;
      int row = (int)m0 + j + 8 * hi;
      int n   = (int)n0 + nt * 16 + col;
      if (mode == 0) {
        outF[(size_t)row * N + n] = val;
      } else {
        int b = row / L_SZ, t = row % L_SZ;
        int h = n / HD_SZ,  d = n % HD_SZ;
        size_t idx = (mode == 2)
            ? ((((size_t)b * H_SZ + h) * L_SZ + t) * HD_SZ + d)
            : ((((size_t)b * H_SZ + h) * HD_SZ + d) * L_SZ + t);
        outB[idx] = f2bf(val);
      }
    }
  }
}

// ---------------------------------------------------------------------------
// Fused causal exponential-decay attention.
//   Q  : [B,H,L,64] bf16 (already scaled by 1/sqrt(64))
//   Kb : [B,H,L,64] bf16
//   VT : [B,H,64,L] bf16 (d-major rows -> contiguous B-fragment loads)
//   RET: [B*L, H*64] bf16
// grid = (L/64, H, B); block = 128 (4 waves); wave owns 16 query rows.
__global__ void retention(const u16* __restrict__ Q, const u16* __restrict__ Kb,
                          const u16* __restrict__ VT,
                          const float* __restrict__ gammas,
                          u16* __restrict__ RET) {
  __shared__ __align__(16) u16 smem[4][16 * 32];  // per-wave 16x32 weight tile

  int w    = threadIdx.x >> 5;
  int lane = threadIdx.x & 31;
  int h = blockIdx.y, b = blockIdx.z;
  unsigned t0 = blockIdx.x * 64 + w * 16;

  const u16* qh  = Q  + ((size_t)b * H_SZ + h) * L_SZ * HD_SZ;
  const u16* kh  = Kb + ((size_t)b * H_SZ + h) * L_SZ * HD_SZ;
  const u16* vth = VT + ((size_t)b * H_SZ + h) * HD_SZ * L_SZ;
  float lg = __log2f(gammas[h]);

  v16bf qf0 = load_frag(qh, HD_SZ, t0, 0);
  v16bf qf1 = load_frag(qh, HD_SZ, t0, 32);

  v8f acc0 = {}, acc1 = {}, acc2 = {}, acc3 = {};
  int hi = lane >> 4, col = lane & 15;

  for (unsigned s0 = 0; s0 < t0 + 16u; s0 += 32u) {
    // ---- scores for two 16-wide s sub-chunks, decay applied in registers ----
#pragma unroll
    for (unsigned sc = 0; sc < 2; ++sc) {
      unsigned sbase = s0 + sc * 16u;
      v8f c = {};
      c = wmma_bf16(qf0, load_frag(kh, HD_SZ, sbase, 0),  c);
      c = wmma_bf16(qf1, load_frag(kh, HD_SZ, sbase, 32), c);
#pragma unroll
      for (int j = 0; j < 8; ++j) {
        int t = (int)t0 + j + 8 * hi;
        int s = (int)sbase + col;
        int d = t - s;
        float wv = (d >= 0) ? c[j] * exp2f((float)d * lg) : 0.0f;
        smem[w][(j + 8 * hi) * 32 + (int)sc * 16 + col] = f2bf(wv);
      }
    }
    // DS ops are in-order per wave; wait for the stores, then re-read the
    // 16x32 weight tile in A-fragment layout.
    asm volatile("s_wait_dscnt 0" ::: "memory");
    v16bf wf = lds_frag(smem[w], 0);
    // ---- retained(16x64) += weights(16x32) @ V(32x64) ----
    acc0 = wmma_bf16(wf, load_frag(vth, L_SZ,  0, s0), acc0);
    acc1 = wmma_bf16(wf, load_frag(vth, L_SZ, 16, s0), acc1);
    acc2 = wmma_bf16(wf, load_frag(vth, L_SZ, 32, s0), acc2);
    acc3 = wmma_bf16(wf, load_frag(vth, L_SZ, 48, s0), acc3);
  }

  v8f accs[4] = {acc0, acc1, acc2, acc3};
#pragma unroll
  for (int nt = 0; nt < 4; ++nt)
#pragma unroll
    for (int j = 0; j < 8; ++j) {
      int t = (int)t0 + j + 8 * hi;
      int n = h * HD_SZ + nt * 16 + col;
      RET[((size_t)b * L_SZ + t) * D_SZ + n] = f2bf(accs[nt][j]);
    }
}

// ---------------------------------------------------------------------------
extern "C" void kernel_launch(void* const* d_in, const int* in_sizes, int n_in,
                              void* d_out, int out_size, void* d_ws, size_t ws_size,
                              hipStream_t stream) {
  (void)in_sizes; (void)n_in; (void)out_size; (void)ws_size;
  const float* x  = (const float*)d_in[0];
  const float* wq = (const float*)d_in[1];
  const float* wk = (const float*)d_in[2];
  const float* wv = (const float*)d_in[3];
  const float* wo = (const float*)d_in[4];
  const float* gm = (const float*)d_in[5];

  char* ws = (char*)d_ws;
  size_t off = 0;
  auto alloc = [&](size_t elems) {
    u16* p = (u16*)(ws + off);
    off += elems * sizeof(u16);
    return p;
  };
  u16* XB   = alloc((size_t)M_SZ * D_SZ);
  u16* WQB  = alloc((size_t)D_SZ * D_SZ);
  u16* WKB  = alloc((size_t)D_SZ * D_SZ);
  u16* WVB  = alloc((size_t)D_SZ * D_SZ);
  u16* WOB  = alloc((size_t)D_SZ * D_SZ);
  u16* QB   = alloc((size_t)M_SZ * D_SZ);
  u16* KB   = alloc((size_t)M_SZ * D_SZ);
  u16* VTB  = alloc((size_t)M_SZ * D_SZ);
  u16* RETB = alloc((size_t)M_SZ * D_SZ);

  const int nX4 = (M_SZ * D_SZ) / 4;
  const int nW4 = (D_SZ * D_SZ) / 4;
  cast_f32_bf16_v4<<<(nX4 + 255) / 256, 256, 0, stream>>>(x,  XB,  nX4);
  cast_f32_bf16_v4<<<(nW4 + 255) / 256, 256, 0, stream>>>(wq, WQB, nW4);
  cast_f32_bf16_v4<<<(nW4 + 255) / 256, 256, 0, stream>>>(wk, WKB, nW4);
  cast_f32_bf16_v4<<<(nW4 + 255) / 256, 256, 0, stream>>>(wv, WVB, nW4);
  cast_f32_bf16_v4<<<(nW4 + 255) / 256, 256, 0, stream>>>(wo, WOB, nW4);

  dim3 gGemm(M_SZ / 128, D_SZ / 64);  // 32 x 16 blocks, 256 threads each
  gemm_bf16<<<gGemm, 256, 0, stream>>>(XB, WQB, nullptr, QB,  M_SZ, D_SZ, D_SZ, 0.125f, 2);
  gemm_bf16<<<gGemm, 256, 0, stream>>>(XB, WKB, nullptr, KB,  M_SZ, D_SZ, D_SZ, 1.0f,   2);
  gemm_bf16<<<gGemm, 256, 0, stream>>>(XB, WVB, nullptr, VTB, M_SZ, D_SZ, D_SZ, 1.0f,   3);

  dim3 gRet(L_SZ / 64, H_SZ, B_SZ);  // 32 x 16 x 2
  retention<<<gRet, 128, 0, stream>>>(QB, KB, VTB, gm, RETB);

  gemm_bf16<<<gGemm, 256, 0, stream>>>(RETB, WOB, (float*)d_out, nullptr,
                                       M_SZ, D_SZ, D_SZ, 1.0f, 0);
}
